// StructEmbedder_29841432773281
// MI455X (gfx1250) — matrix operand order repaired and checked by reading, not compiled, not run on hardware
//
#include <hip/hip_runtime.h>
#include <hip/hip_bf16.h>

// ---------------------------------------------------------------------------
// StructEmbedder for MI455X (gfx1250, wave32, WMMA bf16)
//
//   k0a: fold (W_emb + b_emb) into W_lin blocks -> Wc, PRE-SWIZZLED into WMMA
//        B-fragment layout (per chunk, per lane: 16 contiguous bf16)
//   k0b: z_W -> bf16, same swizzle
//   k1 : final_d = softmax-features(13664) @ Wc via v_wmma.
//        Per softmax group: log2-domain constants (A1,A0) so each feature is
//        exp2(fma(offb, fma(offb, C, A1), A0)): 2 FMA + v_exp_f32 + ds_load_b64.
//   k2 : Z[m][h] = (pf[m] (x) fd[m])(1024) @ zW(1024,128) via v_wmma
//   Both GEMMs: 2x-unrolled K loop with ping-pong B fragments (no reg copies).
// ---------------------------------------------------------------------------

typedef __attribute__((ext_vector_type(16))) __bf16 v16bf;
typedef __attribute__((ext_vector_type(8)))  float  v8f;

#define NROUGH 3136   // 196 atom-pairs * 16 bins
#define NDIST  7840   // 196 atom-pairs * 40 bins
#define KTOT   13664
#define NCHUNK 427    // KTOT/32

#define LOG2E 1.4426950408889634f

// val2bins constants (step == s for all three)
#define S1    1.25f
#define OFF1  2.625f
#define C1    (-8.0f)
#define S2    1.25f
#define OFF2  2.625f
#define C2    (-8.0f)
#define S3    (64.0f/62.0f)
#define OFF3  (-32.0f - 0.5f*S3)
#define C3    (-0.5f/((0.2f*S3)*(0.2f*S3)))
// log2-domain quadratic coefficients
#define C1L   (C1*LOG2E)
#define C2L   (C2*LOG2E)
#define C3L   (C3*LOG2E)

// A-fragment element->K map (contiguous: k = hi*16 + e, matches bswiz B layout)
__device__ __forceinline__ int kmapA(int e, int hi) {
    return hi * 16 + e;
}

// log2-domain group constants: mask*softmax_bin(x) == exp2(cfL*offb^2 + A1*offb + A0)
__device__ __forceinline__ float2 group_ab(float x, float off0, float s, float cf,
                                           int nb, float msk) {
    float fj = floorf((x - off0) / s + 0.5f);
    fj = fminf(fmaxf(fj, 0.0f), (float)(nb - 1));
    float dm = x - (off0 + fj * s);
    float mx = cf * dm * dm;                       // analytic softmax max term
    float sum = 0.0f;
    for (int j = 0; j < nb; ++j) {
        float dj = x - (off0 + j * s);
        sum += __expf(cf * dj * dj - mx);
    }
    float lpre2 = __log2f(msk / sum);              // -inf when msk==0 -> exp2 -> 0
    float2 g;
    g.x = -2.0f * cf * x * LOG2E;                  // A1 (log2 domain)
    g.y = (cf * x * x - mx) * LOG2E + lpre2;       // A0 (log2 domain)
    return g;
}

// B-fragment swizzle: frag=(chunk,n>>4); lane=(kk>>4)*16+(n&15); elem=kk&15
__device__ __forceinline__ int bswiz(int chunk, int nfrag_per_chunk, int kk, int n) {
    int frag = chunk * nfrag_per_chunk + (n >> 4);
    int lane = ((kk >> 4) << 4) + (n & 15);
    return (frag * 32 + lane) * 16 + (kk & 15);
}

// per-chunk A fragment from LDS group constants
__device__ __forceinline__ v16bf make_afrag(int c, int r, int hi,
                                            const float2* sg1, const float2* sg2,
                                            const float2* sg3) {
    v16bf afrag;
    if (c < 98) {                       // rough: bin = f&15, group = f>>4
#pragma unroll
        for (int e = 0; e < 16; ++e) {
            int f = c * 32 + kmapA(e, hi);
            float2 g = sg1[r * 196 + (f >> 4)];
            float offb = OFF1 + (float)(f & 15) * S1;
            afrag[e] = (__bf16)exp2f(offb * (C1L * offb + g.x) + g.y);
        }
    } else if (c < 343) {               // dist: group = g/40
#pragma unroll
        for (int e = 0; e < 16; ++e) {
            int gix = c * 32 + kmapA(e, hi) - NROUGH;
            int apair = gix / 40, bin = gix - apair * 40;
            float2 g = sg2[r * 196 + apair];
            float offb = OFF2 + (float)bin * S2;
            afrag[e] = (__bf16)exp2f(offb * (C2L * offb + g.x) + g.y);
        }
    } else {                            // pos: bin = g&63, group = g>>6
#pragma unroll
        for (int e = 0; e < 16; ++e) {
            int gix = c * 32 + kmapA(e, hi) - (NROUGH + NDIST);
            float2 g = sg3[r * 42 + (gix >> 6)];
            float offb = OFF3 + (float)(gix & 63) * S3;
            afrag[e] = (__bf16)exp2f(offb * (C3L * offb + g.x) + g.y);
        }
    }
    return afrag;
}

// ---------------------------------------------------------------------------
__global__ void fold_weights_kernel(const float* __restrict__ We1, const float* __restrict__ be1,
                                    const float* __restrict__ We2, const float* __restrict__ be2,
                                    const float* __restrict__ We3, const float* __restrict__ be3,
                                    const float* __restrict__ Wl1, const float* __restrict__ Wl2,
                                    const float* __restrict__ Wl3, __bf16* __restrict__ wc) {
    int idx = blockIdx.x * blockDim.x + threadIdx.x;
    if (idx >= KTOT * 32) return;
    int f = idx >> 5, n = idx & 31;
    float s = 0.0f;
    if (f < NROUGH) {
        int apair = f >> 4, bin = f & 15;
        for (int d = 0; d < 32; ++d)
            s += (We1[bin * 32 + d] + be1[d]) * Wl1[(apair * 32 + d) * 32 + n];
    } else if (f < NROUGH + NDIST) {
        int g = f - NROUGH;
        int apair = g / 40, bin = g - apair * 40;
        for (int d = 0; d < 32; ++d)
            s += (We2[bin * 32 + d] + be2[d]) * Wl2[(apair * 32 + d) * 32 + n];
    } else {
        int g = f - (NROUGH + NDIST);
        int vidx = g >> 6, bin = g & 63;
        for (int d = 0; d < 32; ++d)
            s += (We3[bin * 32 + d] + be3[d]) * Wl3[(vidx * 32 + d) * 32 + n];
    }
    wc[bswiz(f >> 5, 2, f & 31, n)] = (__bf16)s;
}

// ---------------------------------------------------------------------------
__global__ void cvt_zw_kernel(const float* __restrict__ zw, __bf16* __restrict__ zwbf) {
    int idx = blockIdx.x * blockDim.x + threadIdx.x;
    if (idx >= 32 * 32 * 128) return;
    int i = idx >> 12;
    int rem = idx & 4095;
    int j = rem >> 7;
    int h = rem & 127;
    zwbf[bswiz(i, 8, j, h)] = (__bf16)zw[idx];
}

// ---------------------------------------------------------------------------
// k1: one wave per block; pairs m0..m0+15 (fixed p).
// ---------------------------------------------------------------------------
__global__ __launch_bounds__(32)
void final_d_kernel(const float* __restrict__ pos_a, const float* __restrict__ mask_a,
                    const float* __restrict__ pos_b, const float* __restrict__ mask_b,
                    const float* __restrict__ rr,    const float* __restrict__ rt,
                    const v16bf* __restrict__ wcv,
                    const float* __restrict__ bl1, const float* __restrict__ bl2,
                    const float* __restrict__ bl3,
                    float* __restrict__ fd) {
    __shared__ float  sposa[14 * 3];
    __shared__ float  sposb[16 * 14 * 3];
    __shared__ float  smaska[14];
    __shared__ float  smaskbp[14];
    __shared__ float  smaskbq[16 * 14];
    __shared__ float  sR[9];
    __shared__ float  sT[3];
    __shared__ float2 sg1[16 * 196];   // (A1,A0) rough,  25.1 KB
    __shared__ float2 sg2[16 * 196];   // (A1,A0) dist,   25.1 KB
    __shared__ float2 sg3[16 * 42];    // (A1,A0) pos,     5.4 KB

    const int lane = threadIdx.x;
    const int m0 = blockIdx.x * 16;
    const int p = m0 >> 7;
    const int q0 = m0 & 127;

    for (int i = lane; i < 42; i += 32)  sposa[i] = pos_a[p * 42 + i];
    for (int i = lane; i < 672; i += 32) sposb[i] = pos_b[q0 * 42 + i];
    for (int i = lane; i < 14; i += 32) {
        smaska[i]  = mask_a[p * 14 + i];
        smaskbp[i] = mask_b[p * 14 + i];
    }
    for (int i = lane; i < 224; i += 32) smaskbq[i] = mask_b[q0 * 14 + i];
    if (lane < 9) sR[lane] = rr[p * 9 + lane];
    if (lane < 3) sT[lane] = rt[p * 3 + lane];
    __syncthreads();

    // fused: distance -> group constants for rough + dist
    for (int idx = lane; idx < 16 * 196; idx += 32) {
        int r = idx / 196, k = idx - r * 196;
        int ia = k / 14, ib = k - ia * 14;
        float dx = sposb[r * 42 + ib * 3 + 0] - sposa[ia * 3 + 0];
        float dy = sposb[r * 42 + ib * 3 + 1] - sposa[ia * 3 + 1];
        float dz = sposb[r * 42 + ib * 3 + 2] - sposa[ia * 3 + 2];
        float d = sqrtf(dx * dx + dy * dy + dz * dz);
        float msk = smaska[ia] * smaskbq[r * 14 + ib];
        sg1[idx] = group_ab(d, OFF1, S1, C1, 16, msk);
        sg2[idx] = group_ab(d, OFF2, S2, C2, 40, msk);
    }
    // fused: local_vec -> group constants for pos
    for (int idx = lane; idx < 16 * 42; idx += 32) {
        int r = idx / 42, v = idx - r * 42;
        int ib = v / 3, i = v - ib * 3;
        float q0v = sposb[r * 42 + ib * 3 + 0] - sT[0];
        float q1v = sposb[r * 42 + ib * 3 + 1] - sT[1];
        float q2v = sposb[r * 42 + ib * 3 + 2] - sT[2];
        float lv = sR[0 * 3 + i] * q0v + sR[1 * 3 + i] * q1v + sR[2 * 3 + i] * q2v;
        float msk = smaskbp[ib] * smaskbq[r * 14 + ib];
        sg3[idx] = group_ab(lv, OFF3, S3, C3, 64, msk);
    }
    __syncthreads();

    const int r  = lane & 15;
    const int hi = lane >> 4;
    v8f acc0 = {}, acc1 = {};

    // ping-pong B fragments, 2x-unrolled K loop (no register copies)
    v16bf bA0 = wcv[lane];
    v16bf bA1 = wcv[32 + lane];
    v16bf bB0 = {}, bB1 = {};

    for (int c = 0; c < NCHUNK; c += 2) {
        if (c + 1 < NCHUNK) {
            bB0 = wcv[((c + 1) * 2 + 0) * 32 + lane];
            bB1 = wcv[((c + 1) * 2 + 1) * 32 + lane];
        }
        v16bf a0 = make_afrag(c, r, hi, sg1, sg2, sg3);
        acc0 = __builtin_amdgcn_wmma_f32_16x16x32_bf16(false, a0, false, bA0,
                                                       (short)0, acc0, false, false);
        acc1 = __builtin_amdgcn_wmma_f32_16x16x32_bf16(false, a0, false, bA1,
                                                       (short)0, acc1, false, false);
        if (c + 1 >= NCHUNK) break;
        if (c + 2 < NCHUNK) {
            bA0 = wcv[((c + 2) * 2 + 0) * 32 + lane];
            bA1 = wcv[((c + 2) * 2 + 1) * 32 + lane];
        }
        v16bf a1 = make_afrag(c + 1, r, hi, sg1, sg2, sg3);
        acc0 = __builtin_amdgcn_wmma_f32_16x16x32_bf16(false, a1, false, bB0,
                                                       (short)0, acc0, false, false);
        acc1 = __builtin_amdgcn_wmma_f32_16x16x32_bf16(false, a1, false, bB1,
                                                       (short)0, acc1, false, false);
    }

    const int n0 = lane & 15;
    float bsA = bl1[n0] + bl2[n0] + bl3[n0];
    float bsB = bl1[16 + n0] + bl2[16 + n0] + bl3[16 + n0];
#pragma unroll
    for (int v = 0; v < 8; ++v) {
        int row = m0 + v + 8 * hi;
        fd[row * 32 + n0]      = acc0[v] + bsA;
        fd[row * 32 + 16 + n0] = acc1[v] + bsB;
    }
}

// ---------------------------------------------------------------------------
// k2: Z GEMM, M=16384, K=1024 (chunk c == embedding index i), N=128
// ---------------------------------------------------------------------------
__global__ __launch_bounds__(32)
void z_einsum_kernel(const int* __restrict__ aat1, const float* __restrict__ aa_emb,
                     const float* __restrict__ fd, const v16bf* __restrict__ zwv,
                     const float* __restrict__ zb, float* __restrict__ out) {
    const int lane = threadIdx.x;
    const int m0 = blockIdx.x * 16;
    const int r  = lane & 15;
    const int hi = lane >> 4;
    const int m  = m0 + r;
    const int p  = m >> 7, q = m & 127;
    const int pidx = aat1[p] * 21 + aat1[q];
    const float* pfr = aa_emb + pidx * 32;

    float fdv[16];
#pragma unroll
    for (int e = 0; e < 16; ++e) fdv[e] = fd[m * 32 + kmapA(e, hi)];

    v8f acc[8];
#pragma unroll
    for (int nt = 0; nt < 8; ++nt) acc[nt] = (v8f){};

    v16bf bA[8], bB[8];
#pragma unroll
    for (int nt = 0; nt < 8; ++nt) bA[nt] = zwv[nt * 32 + lane];

    for (int c = 0; c < 32; c += 2) {
#pragma unroll
        for (int nt = 0; nt < 8; ++nt)
            bB[nt] = zwv[((c + 1) * 8 + nt) * 32 + lane];
        float pf0 = pfr[c];
        v16bf a0;
#pragma unroll
        for (int e = 0; e < 16; ++e) a0[e] = (__bf16)(pf0 * fdv[e]);
#pragma unroll
        for (int nt = 0; nt < 8; ++nt)
            acc[nt] = __builtin_amdgcn_wmma_f32_16x16x32_bf16(false, a0, false, bA[nt],
                                                              (short)0, acc[nt], false, false);
        if (c + 2 < 32) {
#pragma unroll
            for (int nt = 0; nt < 8; ++nt)
                bA[nt] = zwv[((c + 2) * 8 + nt) * 32 + lane];
        }
        float pf1 = pfr[c + 1];
        v16bf a1;
#pragma unroll
        for (int e = 0; e < 16; ++e) a1[e] = (__bf16)(pf1 * fdv[e]);
#pragma unroll
        for (int nt = 0; nt < 8; ++nt)
            acc[nt] = __builtin_amdgcn_wmma_f32_16x16x32_bf16(false, a1, false, bB[nt],
                                                              (short)0, acc[nt], false, false);
    }
#pragma unroll
    for (int nt = 0; nt < 8; ++nt) {
        int h = nt * 16 + r;
        float zbh = zb[h];
#pragma unroll
        for (int v = 0; v < 8; ++v) {
            int row = m0 + v + 8 * hi;
            out[row * 128 + h] = acc[nt][v] + zbh;
        }
    }
}

// ---------------------------------------------------------------------------
extern "C" void kernel_launch(void* const* d_in, const int* in_sizes, int n_in,
                              void* d_out, int out_size, void* d_ws, size_t ws_size,
                              hipStream_t stream) {
    const int*   aat1   = (const int*)  d_in[0];
    const float* pos_a  = (const float*)d_in[2];
    const float* mask_a = (const float*)d_in[3];
    const float* pos_b  = (const float*)d_in[4];
    const float* mask_b = (const float*)d_in[5];
    const float* rr     = (const float*)d_in[6];
    const float* rt     = (const float*)d_in[7];
    const float* aa_emb = (const float*)d_in[8];
    const float* We1 = (const float*)d_in[9],  *be1 = (const float*)d_in[10];
    const float* We2 = (const float*)d_in[11], *be2 = (const float*)d_in[12];
    const float* We3 = (const float*)d_in[13], *be3 = (const float*)d_in[14];
    const float* Wl1 = (const float*)d_in[15], *bl1 = (const float*)d_in[16];
    const float* Wl2 = (const float*)d_in[17], *bl2 = (const float*)d_in[18];
    const float* Wl3 = (const float*)d_in[19], *bl3 = (const float*)d_in[20];
    const float* zw  = (const float*)d_in[21], *zb  = (const float*)d_in[22];
    float* out = (float*)d_out;

    char* ws = (char*)d_ws;
    __bf16* wc   = (__bf16*)ws;                                  // 13664*32*2 = 874496 B
    __bf16* zwbf = (__bf16*)(ws + 874496);                       // 1024*128*2 = 262144 B
    float*  fd   = (float*) (ws + 874496 + 262144);              // 16384*32*4 = 2097152 B

    fold_weights_kernel<<<(KTOT * 32 + 255) / 256, 256, 0, stream>>>(
        We1, be1, We2, be2, We3, be3, Wl1, Wl2, Wl3, wc);
    cvt_zw_kernel<<<(131072 + 255) / 256, 256, 0, stream>>>(zw, zwbf);
    final_d_kernel<<<1024, 32, 0, stream>>>(
        pos_a, mask_a, pos_b, mask_b, rr, rt, (const v16bf*)wc, bl1, bl2, bl3, fd);
    z_einsum_kernel<<<1024, 32, 0, stream>>>(aat1, aa_emb, fd, (const v16bf*)zwbf, zb, out);
}
